// AttentionModule_8100308321084
// MI455X (gfx1250) — matrix operand order repaired and checked by reading, not compiled, not run on hardware
//
#include <hip/hip_runtime.h>
#include <hip/hip_bf16.h>

// ---------------------------------------------------------------------------
// Problem constants (from reference): B=32, NC=128, EMB=256, H=W=128, S=64
// ---------------------------------------------------------------------------
#define NB   32
#define NC   128
#define EMB  256
#define SEQ  64
#define HW   16384        // 128*128

typedef __attribute__((ext_vector_type(16))) _Float16 v16h;
typedef __attribute__((ext_vector_type(8)))  _Float16 v8h;
typedef __attribute__((ext_vector_type(4)))  _Float16 v4h;
typedef __attribute__((ext_vector_type(8)))  float    v8f;

// Padded LDS row strides (in halves / words) to avoid bank conflicts.
#define WPCS_STRIDE 72    // wp [c][s] rows (64 + 8 pad halves)
#define WPSC_STRIDE 136   // wp [s][c] rows (128 + 8 pad halves)
#define IMG_STRIDE  136   // img tile [p][c] rows (128 + 8 pad halves)
#define AT16_STRIDE 72    // attn f16 [p][s] rows (64 + 8 pad halves)
#define AT32_STRIDE 20    // attn f32 [s][p] rows (16 + 4 pad words, 16B aligned)
#define WK_STRIDE   264   // kernel1: [c][e]/[s][e] rows (256 + 8 pad halves)

// gfx1250 async global->LDS path. Builtin signature (from hipcc diagnostic):
//   void __builtin_amdgcn_global_load_async_to_lds_b128(
//       int __vector(4) AS(1)* src, int __vector(4) AS(3)* dst,
//       imm int offset, imm int cpol)
#if __has_builtin(__builtin_amdgcn_global_load_async_to_lds_b128) && \
    __has_builtin(__builtin_amdgcn_s_wait_asynccnt)
#define USE_ASYNC_LDS 1
typedef int b128_t __attribute__((vector_size(16)));
typedef b128_t __attribute__((address_space(1)))* gas_ptr;
typedef b128_t __attribute__((address_space(3)))* las_ptr;
#else
#define USE_ASYNC_LDS 0
#endif

// Build a v16h fragment from two 8-half (b128) LDS chunks.
__device__ __forceinline__ v16h ld_frag(const _Float16* p0, const _Float16* p1) {
  v8h lo = *(const v8h*)p0;
  v8h hi = *(const v8h*)p1;
  return __builtin_shufflevector(lo, hi, 0,1,2,3,4,5,6,7,8,9,10,11,12,13,14,15);
}

__device__ __forceinline__ v8f wmma_f16(v16h a, v16h b, v8f c) {
  return __builtin_amdgcn_wmma_f32_16x16x32_f16(false, a, false, b, (short)0, c,
                                                false, false);
}

// ---------------------------------------------------------------------------
// Kernel 1: wp[b] = W(128x256) @ words[b](256x64) + bias, stored as f16 in
// BOTH [c][s] and [s][c] layouts in workspace (so kernel 2 never transposes).
// One block per batch; wave m owns channel tile m*16..m*16+15.
// ---------------------------------------------------------------------------
__global__ __launch_bounds__(256) void wp_kernel(
    const float* __restrict__ Wg, const float* __restrict__ bias,
    const float* __restrict__ words,
    _Float16* __restrict__ wp_cs_g, _Float16* __restrict__ wp_sc_g)
{
  __shared__ __align__(16) _Float16 Wl[NC * WK_STRIDE];     // [c][e]
  __shared__ __align__(16) _Float16 wordsl[SEQ * WK_STRIDE];// [s][e]

  const int b   = blockIdx.x;
  const int tid = threadIdx.x;

  // Stage W (128x256 f32) -> f16 LDS [c][e]; b128 global loads, b64 LDS stores.
  #pragma unroll
  for (int i = 0; i < 32; ++i) {
    int f4 = (tid + 256 * i) * 4;           // flat float index, multiple of 4
    int c = f4 >> 8, e = f4 & 255;
    float4 v = *(const float4*)&Wg[f4];
    v4h h; h[0] = (_Float16)v.x; h[1] = (_Float16)v.y;
           h[2] = (_Float16)v.z; h[3] = (_Float16)v.w;
    *(v4h*)&Wl[c * WK_STRIDE + e] = h;
  }
  // Stage words[b] (256x64 f32, [e][s]) -> f16 LDS [s][e] (transposed).
  const float* wb = words + (size_t)b * EMB * SEQ;
  #pragma unroll
  for (int i = 0; i < 16; ++i) {
    int f4 = (tid + 256 * i) * 4;           // 4 consecutive s, same e
    int e = f4 >> 6, s = f4 & 63;
    float4 v = *(const float4*)&wb[f4];
    wordsl[(s + 0) * WK_STRIDE + e] = (_Float16)v.x;
    wordsl[(s + 1) * WK_STRIDE + e] = (_Float16)v.y;
    wordsl[(s + 2) * WK_STRIDE + e] = (_Float16)v.z;
    wordsl[(s + 3) * WK_STRIDE + e] = (_Float16)v.w;
  }
  __syncthreads();

  const int wave = tid >> 5, lane = tid & 31;
  const int lrow = lane & 15, hsel = lane >> 4;
  const int m = wave;                        // channel tile

  v8f acc0 = {0,0,0,0,0,0,0,0}, acc1 = acc0, acc2 = acc0, acc3 = acc0;
  #pragma unroll
  for (int k = 0; k < 8; ++k) {              // K = EMB = 8 * 32
    int kb = k * 32;
    // A fragment (16x32): lane holds M=c, K-run of 16 (kb + hsel*16)
    const _Float16* ar = &Wl[(m * 16 + lrow) * WK_STRIDE + kb + hsel * 16];
    v16h a = ld_frag(ar, ar + 8);
    // B fragments (32x16): lane holds N=s, K-run of 16 (kb + hsel*16)
    const _Float16* b0 = &wordsl[(0 * 16 + lrow) * WK_STRIDE + kb + hsel * 16];
    const _Float16* b1 = &wordsl[(1 * 16 + lrow) * WK_STRIDE + kb + hsel * 16];
    const _Float16* b2 = &wordsl[(2 * 16 + lrow) * WK_STRIDE + kb + hsel * 16];
    const _Float16* b3 = &wordsl[(3 * 16 + lrow) * WK_STRIDE + kb + hsel * 16];
    acc0 = wmma_f16(a, ld_frag(b0, b0 + 8), acc0);
    acc1 = wmma_f16(a, ld_frag(b1, b1 + 8), acc1);
    acc2 = wmma_f16(a, ld_frag(b2, b2 + 8), acc2);
    acc3 = wmma_f16(a, ld_frag(b3, b3 + 8), acc3);
  }

  // Add bias, convert to f16, write both layouts.
  _Float16* cs = wp_cs_g + (size_t)b * NC * SEQ;
  _Float16* sc = wp_sc_g + (size_t)b * NC * SEQ;
  #pragma unroll
  for (int r = 0; r < 8; ++r) {
    int c = m * 16 + r + 8 * hsel;
    float bc = bias[c];
    #pragma unroll
    for (int t = 0; t < 4; ++t) {
      int s = t * 16 + lrow;
      float v = (t == 0 ? acc0[r] : t == 1 ? acc1[r] : t == 2 ? acc2[r] : acc3[r]) + bc;
      _Float16 h = (_Float16)v;
      cs[c * SEQ + s] = h;
      sc[s * NC + c]  = h;
    }
  }
}

// ---------------------------------------------------------------------------
// Kernel 2: fused attention. grid = (HW/128, B); 8 waves; wave = 16 pixels.
//   scores(16x64) = img(16x128) @ wp(128x64)  * 1/sqrt(128)   [WMMA f16]
//   masked softmax over S (f32, 16-lane shfl reductions)
//   ww(128x16)    = wp(128x64) @ attn^T(64x16)                [WMMA f16]
// wp staging uses gfx1250 async global->LDS (b128) when available; image tile
// is prefetched (global_prefetch_b8) and overlaps the async wp transfer.
// ---------------------------------------------------------------------------
__global__ __launch_bounds__(256) void attn_main(
    const float* __restrict__ images, const int* __restrict__ mask,
    const _Float16* __restrict__ wp_cs_g, const _Float16* __restrict__ wp_sc_g,
    float* __restrict__ out_ww, float* __restrict__ out_attn)
{
  __shared__ __align__(16) _Float16 wp_cs[NC * WPCS_STRIDE];   // [c][s]
  __shared__ __align__(16) _Float16 wp_sc[SEQ * WPSC_STRIDE];  // [s][c]
  __shared__ int mask_s[SEQ];
  // Per-wave scratch: phase1 img tile [p][c] f16; phase2 reused as attn f16 [p][s]
  __shared__ __align__(16) _Float16 wbuf[8][16 * IMG_STRIDE];
  __shared__ __align__(16) float    at32[8][SEQ * AT32_STRIDE]; // attn f32 [s][p]

  const int b   = blockIdx.y;
  const int tid = threadIdx.x;

  // ---- Prefetch this block's image tile (128 ch x 128 px = 64 KB) ---------
  {
    const float* pf = images + (size_t)b * NC * HW + blockIdx.x * 128;
    int c = tid >> 1, half = tid & 1;
    __builtin_prefetch(pf + (size_t)c * HW + half * 64, 0, 3);
  }

  // ---- Stage wp (both layouts) from L2-resident workspace -----------------
  const _Float16* gcs = wp_cs_g + (size_t)b * NC * SEQ;
  const _Float16* gsc = wp_sc_g + (size_t)b * NC * SEQ;
#if USE_ASYNC_LDS
  #pragma unroll
  for (int i = 0; i < 4; ++i) {
    int h8 = (tid + 256 * i) * 8;            // flat half index, multiple of 8
    int c = h8 >> 6, s = h8 & 63;
    __builtin_amdgcn_global_load_async_to_lds_b128(
        (gas_ptr)&gcs[h8], (las_ptr)&wp_cs[c * WPCS_STRIDE + s], 0, 0);
  }
  #pragma unroll
  for (int i = 0; i < 4; ++i) {
    int h8 = (tid + 256 * i) * 8;
    int s = h8 >> 7, c = h8 & 127;
    __builtin_amdgcn_global_load_async_to_lds_b128(
        (gas_ptr)&gsc[h8], (las_ptr)&wp_sc[s * WPSC_STRIDE + c], 0, 0);
  }
#else
  #pragma unroll
  for (int i = 0; i < 4; ++i) {
    int h8 = (tid + 256 * i) * 8;
    int c = h8 >> 6, s = h8 & 63;
    *(v8h*)&wp_cs[c * WPCS_STRIDE + s] = *(const v8h*)&gcs[h8];
  }
  #pragma unroll
  for (int i = 0; i < 4; ++i) {
    int h8 = (tid + 256 * i) * 8;
    int s = h8 >> 7, c = h8 & 127;
    *(v8h*)&wp_sc[s * WPSC_STRIDE + c] = *(const v8h*)&gsc[h8];
  }
#endif
  if (tid < SEQ) mask_s[tid] = mask[b * SEQ + tid];

  const int wave = tid >> 5, lane = tid & 31;
  const int lrow = lane & 15, hsel = lane >> 4;
  const int pTile = blockIdx.x * 128 + wave * 16;

  // ---- Stage img tile: 128 channels x 16 pixels, f32 -> f16 [p][c] --------
  // (overlaps with the async wp transfer above)
  const float* imgb = images + (size_t)b * NC * HW + pTile;
  _Float16* img_s = &wbuf[wave][0];
  #pragma unroll
  for (int i = 0; i < 16; ++i) {
    int c  = i * 8 + (lane >> 2);
    int p0 = (lane & 3) * 4;
    float4 v = *(const float4*)&imgb[(size_t)c * HW + p0];
    img_s[(p0 + 0) * IMG_STRIDE + c] = (_Float16)v.x;
    img_s[(p0 + 1) * IMG_STRIDE + c] = (_Float16)v.y;
    img_s[(p0 + 2) * IMG_STRIDE + c] = (_Float16)v.z;
    img_s[(p0 + 3) * IMG_STRIDE + c] = (_Float16)v.w;
  }
#if USE_ASYNC_LDS
  __builtin_amdgcn_s_wait_asynccnt(0);
#endif
  __syncthreads();

  // ---- Scores: acc[t](16p x 16s) += img(16x32) @ wp(32x16), K = NC --------
  v8f acc0 = {0,0,0,0,0,0,0,0}, acc1 = acc0, acc2 = acc0, acc3 = acc0;
  #pragma unroll
  for (int k = 0; k < 4; ++k) {
    int kb = k * 32;
    const _Float16* ar = &img_s[lrow * IMG_STRIDE + kb + hsel * 16];
    v16h a = ld_frag(ar, ar + 8);
    const _Float16* b0 = &wp_sc[(0 * 16 + lrow) * WPSC_STRIDE + kb + hsel * 16];
    const _Float16* b1 = &wp_sc[(1 * 16 + lrow) * WPSC_STRIDE + kb + hsel * 16];
    const _Float16* b2 = &wp_sc[(2 * 16 + lrow) * WPSC_STRIDE + kb + hsel * 16];
    const _Float16* b3 = &wp_sc[(3 * 16 + lrow) * WPSC_STRIDE + kb + hsel * 16];
    acc0 = wmma_f16(a, ld_frag(b0, b0 + 8), acc0);
    acc1 = wmma_f16(a, ld_frag(b1, b1 + 8), acc1);
    acc2 = wmma_f16(a, ld_frag(b2, b2 + 8), acc2);
    acc3 = wmma_f16(a, ld_frag(b3, b3 + 8), acc3);
  }

  // ---- Masked softmax over S=64 (row p lives across the 16-lane group) ----
  const float scale = 0.08838834764831845f;  // 1/sqrt(128)
  const int m0 = mask_s[0 * 16 + lrow], m1 = mask_s[1 * 16 + lrow];
  const int m2 = mask_s[2 * 16 + lrow], m3 = mask_s[3 * 16 + lrow];
  const float NEGINF = -__builtin_inff();
  float*     a32 = &at32[wave][0];
  _Float16*  a16 = img_s;                    // reuse img region as attn f16 [p][s]
  #pragma unroll
  for (int r = 0; r < 8; ++r) {
    float s0 = m0 ? acc0[r] * scale : NEGINF;
    float s1 = m1 ? acc1[r] * scale : NEGINF;
    float s2 = m2 ? acc2[r] * scale : NEGINF;
    float s3 = m3 ? acc3[r] * scale : NEGINF;
    float mx = fmaxf(fmaxf(s0, s1), fmaxf(s2, s3));
    mx = fmaxf(mx, __shfl_xor(mx, 1, 16));
    mx = fmaxf(mx, __shfl_xor(mx, 2, 16));
    mx = fmaxf(mx, __shfl_xor(mx, 4, 16));
    mx = fmaxf(mx, __shfl_xor(mx, 8, 16));
    float e0 = m0 ? __expf(s0 - mx) : 0.f;
    float e1 = m1 ? __expf(s1 - mx) : 0.f;
    float e2 = m2 ? __expf(s2 - mx) : 0.f;
    float e3 = m3 ? __expf(s3 - mx) : 0.f;
    float sum = e0 + e1 + e2 + e3;
    sum += __shfl_xor(sum, 1, 16);
    sum += __shfl_xor(sum, 2, 16);
    sum += __shfl_xor(sum, 4, 16);
    sum += __shfl_xor(sum, 8, 16);
    float inv = 1.0f / sum;
    float w0 = e0 * inv, w1 = e1 * inv, w2 = e2 * inv, w3 = e3 * inv;
    int prel = r + 8 * hsel;
    a32[(0 * 16 + lrow) * AT32_STRIDE + prel] = w0;
    a32[(1 * 16 + lrow) * AT32_STRIDE + prel] = w1;
    a32[(2 * 16 + lrow) * AT32_STRIDE + prel] = w2;
    a32[(3 * 16 + lrow) * AT32_STRIDE + prel] = w3;
    a16[prel * AT16_STRIDE + (0 * 16 + lrow)] = (_Float16)w0;
    a16[prel * AT16_STRIDE + (1 * 16 + lrow)] = (_Float16)w1;
    a16[prel * AT16_STRIDE + (2 * 16 + lrow)] = (_Float16)w2;
    a16[prel * AT16_STRIDE + (3 * 16 + lrow)] = (_Float16)w3;
  }
  __syncthreads();

  // ---- Coalesced attn f32 writeout: out_attn[b][s][p] ---------------------
  const size_t attnBase = (size_t)b * SEQ * HW + pTile;
  #pragma unroll
  for (int i = 0; i < 8; ++i) {
    int s  = i * 8 + (lane >> 2);
    int p0 = (lane & 3) * 4;
    float4 v = *(const float4*)&a32[s * AT32_STRIDE + p0];  // 16B-aligned: b128
    *(float4*)&out_attn[attnBase + (size_t)s * HW + p0] = v;
  }

  // ---- ww: for each channel tile m: ww(16c x 16p) = wp(16x64) @ attn^T ----
  #pragma unroll
  for (int m = 0; m < 8; ++m) {
    v8f accw = {0,0,0,0,0,0,0,0};
    #pragma unroll
    for (int k2 = 0; k2 < 2; ++k2) {
      int sb = k2 * 32;
      const _Float16* ar = &wp_cs[(m * 16 + lrow) * WPCS_STRIDE + sb + hsel * 16];
      v16h a = ld_frag(ar, ar + 8);
      const _Float16* br = &a16[lrow * AT16_STRIDE + sb + hsel * 16];
      accw = wmma_f16(a, ld_frag(br, br + 8), accw);
    }
    size_t base = ((size_t)b * NC + m * 16 + 8 * hsel) * HW + pTile + lrow;
    #pragma unroll
    for (int r = 0; r < 8; ++r)
      out_ww[base + (size_t)r * HW] = accw[r];
  }
}

// ---------------------------------------------------------------------------
extern "C" void kernel_launch(void* const* d_in, const int* in_sizes, int n_in,
                              void* d_out, int out_size, void* d_ws, size_t ws_size,
                              hipStream_t stream) {
  const float* images = (const float*)d_in[0];   // [B, NC, H, W]
  const float* words  = (const float*)d_in[1];   // [B, EMB, S]
  const int*   maskp  = (const int*)d_in[2];     // [B, S]
  const float* Wg     = (const float*)d_in[3];   // [NC, EMB]
  const float* bias   = (const float*)d_in[4];   // [NC]

  float* out_ww   = (float*)d_out;                         // [B, NC, H, W]
  float* out_attn = out_ww + (size_t)NB * NC * HW;         // [B, S, H, W]

  _Float16* wp_cs_g = (_Float16*)d_ws;                     // [B][c][s]
  _Float16* wp_sc_g = wp_cs_g + (size_t)NB * NC * SEQ;     // [B][s][c]

  wp_kernel<<<NB, 256, 0, stream>>>(Wg, bias, words, wp_cs_g, wp_sc_g);
  attn_main<<<dim3(HW / 128, NB), 256, 0, stream>>>(images, maskp, wp_cs_g,
                                                    wp_sc_g, out_ww, out_attn);
}
